// FRN_70987219468923
// MI455X (gfx1250) — matrix operand order repaired and checked by reading, not compiled, not run on hardware
//
#include <hip/hip_runtime.h>

// ---- problem constants (match reference) ----
#define B_SZ 16384
#define M_SZ 9
#define D_SZ 512

constexpr int ASTR = D_SZ + 8;  // padded A row stride (elems); 1040B, 16B aligned
constexpr int BSTR = 40;        // padded B row stride (elems); 80B, 16B aligned
constexpr int KCH  = 32;        // K per WMMA step

typedef __attribute__((ext_vector_type(16))) __bf16 v16bf;
typedef __attribute__((ext_vector_type(8)))  float  v8f;

struct alignas(16) Q4 { unsigned a, b, c, d; };
struct alignas(8)  Q2 { unsigned a, b; };
union FragU { v16bf v; Q4 q[2]; };

__device__ __forceinline__ unsigned short f2bf(float f) {
    unsigned u = __float_as_uint(f);
    u += 0x7FFFu + ((u >> 16) & 1u);       // round-to-nearest-even
    return (unsigned short)(u >> 16);
}

__device__ __forceinline__ void wait_async0() {
    asm volatile("s_wait_asynccnt 0x0" ::: "memory");
}

// Copy one 64B row chunk global->LDS as 4 async b128s sharing one base pair.
// NOTE: the 24-bit INST_OFFSET is added to BOTH the LDS and global addresses
// (ISA 10, async pseudocode), and both sides are k-contiguous within a row.
__device__ __forceinline__ void async_row64(unsigned l, unsigned g, const void* sbase) {
    asm volatile(
        "global_load_async_to_lds_b128 %0, %1, %2\n\t"
        "global_load_async_to_lds_b128 %0, %1, %2 offset:16\n\t"
        "global_load_async_to_lds_b128 %0, %1, %2 offset:32\n\t"
        "global_load_async_to_lds_b128 %0, %1, %2 offset:48"
        :: "v"(l), "v"(g), "s"(sbase) : "memory");
}

// =====================================================================
// Weight prep: W f32 [m][k][n]  ->  Wt bf16 [m][n][k]  (tiled transpose)
// =====================================================================
__global__ __launch_bounds__(256)
void frn_wprep(const float* __restrict__ W, unsigned short* __restrict__ Wt) {
    __shared__ unsigned short tile[32 * 34];
    const int m  = blockIdx.z;
    const int k0 = blockIdx.y << 5;
    const int n0 = blockIdx.x << 5;
    const int f  = threadIdx.x;

    const float* src = W + ((size_t)m * D_SZ + k0) * D_SZ + n0;
    #pragma unroll
    for (int j = 0; j < 4; ++j) {
        int idx = f + (j << 8);
        int kl = idx >> 5, nl = idx & 31;
        tile[nl * 34 + kl] = f2bf(src[(size_t)kl * D_SZ + nl]);
    }
    __syncthreads();
    unsigned short* dst = Wt + ((size_t)m * D_SZ + n0) * D_SZ + k0;
    #pragma unroll
    for (int j = 0; j < 2; ++j) {
        int idx = f + (j << 8);
        int nl = idx >> 4, kk = (idx & 15) << 1;
        unsigned p = (unsigned)tile[nl * 34 + kk] | ((unsigned)tile[nl * 34 + kk + 1] << 16);
        *(unsigned*)&dst[(size_t)nl * D_SZ + kk] = p;
    }
}

// =====================================================================
// Fused dual-GEMM kernel, 64-row tile.
//  Key identity: (alpha (*) x) @ We = diag(alpha) * (x @ We)
//  -> both GEMMs share A = bf16(x); alpha scales phase-2 OUTPUT rows (f32).
//  Phase 1: acc = x@Wi  -> alphas[b,m] = sum_n sigmoid(acc + bi)   (LDS + global)
//  Phase 2: acc = x@We  -> g = relu(alpha_row*acc + be)            -> out region
// =====================================================================
constexpr int RT = 64;                       // batch rows per workgroup

__global__ __launch_bounds__(256)
void frn_fused(const float* __restrict__ X,
               const unsigned short* __restrict__ Wti,
               const unsigned short* __restrict__ Wte,
               const float* __restrict__ bi, const float* __restrict__ be,
               float* __restrict__ alphas_out, float* __restrict__ gout) {
    __shared__ unsigned short Ash[RT * ASTR];        // 66,560 B
    __shared__ unsigned short Bsh[2][D_SZ * BSTR];   // 81,920 B
    __shared__ float          red[8 * RT];           //  2,048 B
    __shared__ float          alphaS[RT];            //    256 B

    const int tid  = threadIdx.x;
    const int lane = tid & 31;
    const int wave = tid >> 5;
    const int half = lane >> 4;
    const int m    = blockIdx.x / (B_SZ / RT);       // blocks grouped by m -> W[m] hot in L2
    const int b0   = (blockIdx.x % (B_SZ / RT)) * RT;

    // ---- stage A tile (64 x 512) as bf16 once: float4 loads, b64 packed stores ----
    #pragma unroll 4
    for (int i = 0; i < RT * D_SZ / (256 * 4); ++i) {   // 32 iterations
        int idx4 = (tid + (i << 8)) << 2;
        int row = idx4 >> 9, col = idx4 & 511;
        const float* xr = X + ((size_t)(b0 + row) * M_SZ + m) * D_SZ + col;
        float4 xv = *(const float4*)xr;
        Q2 p;
        p.a = (unsigned)f2bf(xv.x) | ((unsigned)f2bf(xv.y) << 16);
        p.b = (unsigned)f2bf(xv.z) | ((unsigned)f2bf(xv.w) << 16);
        *(Q2*)&Ash[row * ASTR + col] = p;
    }

    // ---- async B staging: thread owns rows n=tid and n=tid+256 (64B each) ----
    const unsigned lrow = (unsigned)(size_t)&Bsh[0][0] + (unsigned)tid * (BSTR * 2);
    const unsigned grow = (unsigned)tid * (D_SZ * 2);
    auto stageB = [&](const unsigned short* Wmat, int k0, int buf) {
        unsigned l0 = lrow + (buf ? (unsigned)(D_SZ * BSTR * 2) : 0u);
        unsigned g0 = grow + (unsigned)(k0 * 2);
        async_row64(l0, g0, Wmat);
        async_row64(l0 + 256u * (BSTR * 2), g0 + 256u * (D_SZ * 2), Wmat);
    };

    const int ncol = (wave << 6) + (lane & 15);      // wave covers n in [64w, 64w+64)
    v8f acc[4][4];
    const v8f vzero = {};

    auto gemm_phase = [&](const unsigned short* Wmat) {
        #pragma unroll
        for (int rt = 0; rt < 4; ++rt)
            #pragma unroll
            for (int ct = 0; ct < 4; ++ct) acc[rt][ct] = vzero;
        stageB(Wmat, 0, 0);
        #pragma unroll
        for (int s = 0; s < D_SZ / KCH; ++s) {       // 16 steps, double-buffered
            wait_async0();
            __syncthreads();
            if (s < D_SZ / KCH - 1) stageB(Wmat, (s + 1) << 5, (s + 1) & 1);
            const int k0 = s << 5;

            FragU fa[4];
            #pragma unroll
            for (int rt = 0; rt < 4; ++rt) {
                const unsigned short* ap = &Ash[(rt * 16 + (lane & 15)) * ASTR + k0 + 8 * half];
                fa[rt].q[0] = *(const Q4*)ap;
                fa[rt].q[1] = *(const Q4*)(ap + 16);
            }
            #pragma unroll
            for (int ct = 0; ct < 4; ++ct) {
                FragU fb;
                const unsigned short* bp = &Bsh[s & 1][(ncol + ct * 16) * BSTR + 8 * half];
                fb.q[0] = *(const Q4*)bp;
                fb.q[1] = *(const Q4*)(bp + 16);
                #pragma unroll
                for (int rt = 0; rt < 4; ++rt)
                    acc[rt][ct] = __builtin_amdgcn_wmma_f32_16x16x32_bf16(
                        false, fa[rt].v, false, fb.v, (short)0, acc[rt][ct], false, false);
            }
        }
    };

    // ================= phase 1: x @ Wi -> alphas =================
    gemm_phase(Wti);
    {
        float sums[4][8];
        #pragma unroll
        for (int rt = 0; rt < 4; ++rt)
            #pragma unroll
            for (int r = 0; r < 8; ++r) sums[rt][r] = 0.f;
        #pragma unroll
        for (int ct = 0; ct < 4; ++ct) {
            float bv = bi[m * D_SZ + ncol + ct * 16];
            #pragma unroll
            for (int rt = 0; rt < 4; ++rt)
                #pragma unroll
                for (int r = 0; r < 8; ++r) {
                    float y = acc[rt][ct][r] + bv;
                    sums[rt][r] += 1.0f / (1.0f + __expf(-y));   // |sigmoid|==sigmoid
                }
        }
        #pragma unroll
        for (int rt = 0; rt < 4; ++rt)
            #pragma unroll
            for (int r = 0; r < 8; ++r)
                #pragma unroll
                for (int off = 1; off < 16; off <<= 1)
                    sums[rt][r] += __shfl_xor(sums[rt][r], off, 32);
        if ((lane & 15) == 0) {
            #pragma unroll
            for (int rt = 0; rt < 4; ++rt)
                #pragma unroll
                for (int r = 0; r < 8; ++r)
                    red[wave * RT + rt * 16 + half * 8 + r] = sums[rt][r];
        }
        __syncthreads();
        if (tid < RT) {
            float tot = 0.f;
            #pragma unroll
            for (int w = 0; w < 8; ++w) tot += red[w * RT + tid];
            alphaS[tid] = tot;
            alphas_out[(size_t)(b0 + tid) * M_SZ + m] = tot;
        }
    }

    // ================= phase 2: x @ We -> g =================
    gemm_phase(Wte);   // phase-2 loop barriers also publish alphaS
    #pragma unroll
    for (int ct = 0; ct < 4; ++ct) {
        int n = ncol + ct * 16;
        float bv = be[m * D_SZ + n];
        #pragma unroll
        for (int rt = 0; rt < 4; ++rt)
            #pragma unroll
            for (int r = 0; r < 8; ++r) {
                int row = rt * 16 + r + 8 * half;
                float y = fmaxf(alphaS[row] * acc[rt][ct][r] + bv, 0.f);
                gout[((size_t)(b0 + row) * M_SZ + m) * D_SZ + n] = y;
            }
    }
}

// =====================================================================
// Fallback GEMM (round-1 style, inline f32->bf16 conversion, no scratch)
// =====================================================================
template <int MODE>
__global__ __launch_bounds__(256)
void frn_gemm_inline(const float* __restrict__ X, const float* __restrict__ W,
                     const float* __restrict__ bias, const float* __restrict__ alphas_in,
                     float* __restrict__ out) {
    __shared__ unsigned short Ash[16 * ASTR];
    __shared__ unsigned short Bsh[D_SZ * BSTR];
    __shared__ float          red[8 * 16];

    const int tid  = threadIdx.x;
    const int lane = tid & 31;
    const int wave = tid >> 5;
    const int half = lane >> 4;
    const int m    = blockIdx.x / (B_SZ / 16);
    const int b0   = (blockIdx.x % (B_SZ / 16)) * 16;
    const float* Wm = W + (size_t)m * D_SZ * D_SZ;

    #pragma unroll 4
    for (int j = 0; j < 16; ++j) {
        const int col = 2 * tid;
        const float* xr = X + ((size_t)(b0 + j) * M_SZ + m) * D_SZ;
        float x0 = xr[col], x1 = xr[col + 1];
        if (MODE == 1) {
            float a = alphas_in[(size_t)(b0 + j) * M_SZ + m];
            x0 *= a; x1 *= a;
        }
        unsigned p = (unsigned)f2bf(x0) | ((unsigned)f2bf(x1) << 16);
        *(unsigned*)&Ash[j * ASTR + col] = p;
    }

    v8f acc[4] = {};
    const int ncol = (wave << 6) + (lane & 15);

    for (int k0 = 0; k0 < D_SZ; k0 += KCH) {
        __syncthreads();
        #pragma unroll 8
        for (int j = 0; j < 64; ++j) {
            int idx = tid + (j << 8);
            int kr = idx >> 9, n = idx & 511;
            Bsh[n * BSTR + kr] = f2bf(Wm[(size_t)(k0 + kr) * D_SZ + n]);
        }
        __syncthreads();
        FragU fa;
        const unsigned short* ap = &Ash[(lane & 15) * ASTR + k0 + 8 * half];
        fa.q[0] = *(const Q4*)ap;
        fa.q[1] = *(const Q4*)(ap + 16);
        #pragma unroll
        for (int t = 0; t < 4; ++t) {
            FragU fb;
            const unsigned short* bp = &Bsh[(ncol + t * 16) * BSTR + 8 * half];
            fb.q[0] = *(const Q4*)bp;
            fb.q[1] = *(const Q4*)(bp + 16);
            acc[t] = __builtin_amdgcn_wmma_f32_16x16x32_bf16(
                false, fa.v, false, fb.v, (short)0, acc[t], false, false);
        }
    }

    if (MODE == 0) {
        float s[8];
        #pragma unroll
        for (int r = 0; r < 8; ++r) s[r] = 0.f;
        #pragma unroll
        for (int t = 0; t < 4; ++t) {
            float bv = bias[m * D_SZ + ncol + t * 16];
            #pragma unroll
            for (int r = 0; r < 8; ++r) {
                float y = acc[t][r] + bv;
                s[r] += 1.0f / (1.0f + __expf(-y));
            }
        }
        #pragma unroll
        for (int r = 0; r < 8; ++r)
            #pragma unroll
            for (int off = 1; off < 16; off <<= 1)
                s[r] += __shfl_xor(s[r], off, 32);
        if ((lane & 15) == 0) {
            #pragma unroll
            for (int r = 0; r < 8; ++r) red[wave * 16 + half * 8 + r] = s[r];
        }
        __syncthreads();
        if (tid < 16) {
            float tot = 0.f;
            #pragma unroll
            for (int w = 0; w < 8; ++w) tot += red[w * 16 + tid];
            out[(size_t)(b0 + tid) * M_SZ + m] = tot;
        }
    } else {
        #pragma unroll
        for (int t = 0; t < 4; ++t) {
            int n = ncol + t * 16;
            float bv = bias[m * D_SZ + n];
            #pragma unroll
            for (int r = 0; r < 8; ++r) {
                float y = fmaxf(acc[t][r] + bv, 0.f);
                out[((size_t)(b0 + r + 8 * half) * M_SZ + m) * D_SZ + n] = y;
            }
        }
    }
}

// =====================================================================
// Inter-branch mixing (exact f32): Gram 9x9, cdist, tanh mask, in-place
// =====================================================================
__global__ __launch_bounds__(512)
void frn_inter(float* __restrict__ outg) {
    __shared__ float psum[81];
    __shared__ float affin[81];
    const int tid  = threadIdx.x;      // == d index
    const int lane = tid & 31;
    const size_t base = (size_t)blockIdx.x * (M_SZ * D_SZ);

    float g[M_SZ];
    #pragma unroll
    for (int n = 0; n < M_SZ; ++n) g[n] = outg[base + n * D_SZ + tid];

    if (tid < 81) psum[tid] = 0.f;
    __syncthreads();

    for (int i = 0; i < M_SZ; ++i) {
        for (int j = i; j < M_SZ; ++j) {
            float v = g[i] * g[j];
            #pragma unroll
            for (int off = 1; off < 32; off <<= 1) v += __shfl_xor(v, off, 32);
            if (lane == 0) atomicAdd(&psum[i * 9 + j], v);
        }
    }
    __syncthreads();

    if (tid < 81) {
        int i = tid / 9, j = tid % 9;
        int ii = i < j ? i : j, jj = i < j ? j : i;
        float dot = psum[ii * 9 + jj];
        float d2  = psum[i * 9 + i] + psum[j * 9 + j] - 2.0f * dot;
        d2 = fmaxf(d2, 0.f);
        float dist = (d2 > 0.f) ? sqrtf(d2) : 0.f;
        affin[tid] = (i == j) ? 0.f : tanhf(dist);
    }
    __syncthreads();

    #pragma unroll
    for (int mr = 0; mr < M_SZ; ++mr) {
        float h = 0.f;
        #pragma unroll
        for (int n = 0; n < M_SZ; ++n) h += affin[mr * 9 + n] * g[n];
        outg[base + mr * D_SZ + tid] = 0.5f * g[mr] + 0.5f * h;  // DELTA = 0.5
    }
}

extern "C" void kernel_launch(void* const* d_in, const int* in_sizes, int n_in,
                              void* d_out, int out_size, void* d_ws, size_t ws_size,
                              hipStream_t stream) {
    const float* x  = (const float*)d_in[0];
    const float* Wi = (const float*)d_in[1];
    const float* bi = (const float*)d_in[2];
    const float* We = (const float*)d_in[3];
    const float* be = (const float*)d_in[4];

    float* out    = (float*)d_out;                     // [B, M, D]
    float* alphas = out + (size_t)B_SZ * M_SZ * D_SZ;  // [B, M] (tuple tail)

    const size_t WT_ELEMS = (size_t)M_SZ * D_SZ * D_SZ;
    const size_t need = 2 * WT_ELEMS * sizeof(unsigned short);   // 9.44 MB

    if (ws_size >= need) {
        unsigned short* Wti = (unsigned short*)d_ws;
        unsigned short* Wte = Wti + WT_ELEMS;
        dim3 pg(D_SZ / 32, D_SZ / 32, M_SZ);
        frn_wprep<<<pg, 256, 0, stream>>>(Wi, Wti);
        frn_wprep<<<pg, 256, 0, stream>>>(We, Wte);
        dim3 grid(M_SZ * (B_SZ / RT));   // 2304 WGs, grouped by m
        frn_fused<<<grid, 256, 0, stream>>>(x, Wti, Wte, bi, be, alphas, out);
    } else {
        dim3 grid(M_SZ * (B_SZ / 16));
        frn_gemm_inline<0><<<grid, 256, 0, stream>>>(x, Wi, bi, nullptr, alphas);
        frn_gemm_inline<1><<<grid, 256, 0, stream>>>(x, We, be, alphas, out);
    }
    frn_inter<<<B_SZ, 512, 0, stream>>>(out);
}
